// BertSelfAttention_51290499449146
// MI455X (gfx1250) — matrix operand order repaired and checked by reading, not compiled, not run on hardware
//
#include <hip/hip_runtime.h>
#include <hip/hip_bf16.h>

// ---------------- problem constants ----------------
#define BB   4
#define TT   2048
#define HH   768
#define NHH  12
#define DHH  64
#define KQQ  1024
#define KKVV 1024

typedef __attribute__((ext_vector_type(8)))  __bf16 v8bf;
typedef __attribute__((ext_vector_type(16))) __bf16 v16bf;
typedef __attribute__((ext_vector_type(8)))  float  v8f;

__device__ inline v8f wmma_bf16(v16bf a, v16bf b, v8f c) {
    // D = A(16x32 bf16) * B(32x16 bf16) + C(16x16 f32)
    return __builtin_amdgcn_wmma_f32_16x16x32_bf16(
        /*neg_a=*/false, a, /*neg_b=*/false, b,
        /*c_mod=*/(short)0, c, /*reuse_a=*/false, /*reuse_b=*/false);
}

// A-matrix 16x32 bf16 lane layout used here (contiguous k, not the ISA table order; B matches):
//   lanes 0-15  (half=0): M=lane,    K = {k0+0..15}
//   lanes 16-31 (half=1): M=lane-16, K = {k0+16..31}
__device__ inline v16bf load_a_chunk(const __bf16* row, int k0, int half) {
    union { v16bf v; v8bf h[2]; } u;
    u.h[0] = *(const v8bf*)(row + k0 + 16 * half);
    u.h[1] = *(const v8bf*)(row + k0 + 16 * half + 8);
    return u.v;
}

// ---------------- prep kernels ----------------
__global__ void k_cvt_bf16(const float* __restrict__ s, __bf16* __restrict__ d, int n) {
    int i = blockIdx.x * blockDim.x + threadIdx.x;
    if (i < n) d[i] = (__bf16)s[i];
}

__global__ void k_gmask(const float* __restrict__ amask, const int* __restrict__ kvidx,
                        float* __restrict__ gmask) {
    int i = blockIdx.x * blockDim.x + threadIdx.x;
    if (i >= BB * KKVV) return;
    int b = i / KKVV;
    gmask[i] = amask[b * TT + kvidx[i]];
}

// ---------------- projection GEMM: Y = gather(X)[K,H] * W^T[H,O] + bias ----------------
// grid: (Krows/16, H/64, B), block: 32 (1 wave). One wave -> 16x64 output tile.
__global__ __launch_bounds__(32)
void k_proj(const __bf16* __restrict__ hid, const __bf16* __restrict__ W,
            const float* __restrict__ bias, const int* __restrict__ idx,
            __bf16* __restrict__ out, int Krows) {
    const int lane = threadIdx.x;
    const int c = lane & 15, half = lane >> 4;
    const int b  = blockIdx.z;
    const int m0 = blockIdx.x * 16;
    const int n0 = blockIdx.y * 64;

    const int tok = idx[b * Krows + m0 + c];           // A-row for this lane (M = c)
    const __bf16* grow = hid + ((size_t)b * TT + tok) * HH;

    v8f acc[4];
#pragma unroll
    for (int t = 0; t < 4; ++t) acc[t] = (v8f)(0.0f);

    for (int k0 = 0; k0 < HH; k0 += 32) {
        v16bf a = load_a_chunk(grow, k0, half);
#pragma unroll
        for (int t = 0; t < 4; ++t) {
            // B[k][n] = W[(n0+16t+c)*H + k0+k]; per-lane 16 contiguous bf16 at 16*half
            const __bf16* wrow = W + (size_t)(n0 + t * 16 + c) * HH + k0 + 16 * half;
            v16bf bv = *(const v16bf*)wrow;
            acc[t] = wmma_bf16(a, bv, acc[t]);
        }
    }
    // C/D layout: VGPR r -> row (r + 8*half), column c
#pragma unroll
    for (int t = 0; t < 4; ++t) {
        const int o = n0 + t * 16 + c;
        const float bb = bias[o];
#pragma unroll
        for (int r = 0; r < 8; ++r) {
            const int row = m0 + r + 8 * half;
            out[((size_t)b * Krows + row) * HH + o] = (__bf16)(acc[t][r] + bb);
        }
    }
}

// ---------------- V transpose: vg[b,kv,H] -> vt[b,h,d,kv] ----------------
__global__ void k_vtrans(const __bf16* __restrict__ vg, __bf16* __restrict__ vt) {
    int i = blockIdx.x * blockDim.x + threadIdx.x;
    if (i >= BB * KKVV * HH) return;
    int hh = i % HH;
    int kv = (i / HH) % KKVV;
    int b  = i / (HH * KKVV);
    int h = hh / DHH, d = hh % DHH;
    vt[(((size_t)b * NHH + h) * DHH + d) * KKVV + kv] =
        vg[((size_t)b * KKVV + kv) * HH + hh];
}

// ---------------- background row (non-selected queries): softmax(mask)-weighted V mean ----
__global__ __launch_bounds__(256)
void k_bg(const float* __restrict__ gmask, const __bf16* __restrict__ vg,
          float* __restrict__ bg) {
    __shared__ float red[256];
    const int b = blockIdx.x, tid = threadIdx.x;
    const float* gm = gmask + b * KKVV;

    float mx = -3.0e38f;
    for (int j = tid; j < KKVV; j += 256) mx = fmaxf(mx, gm[j]);
    red[tid] = mx; __syncthreads();
    for (int s = 128; s > 0; s >>= 1) {
        if (tid < s) red[tid] = fmaxf(red[tid], red[tid + s]);
        __syncthreads();
    }
    mx = red[0]; __syncthreads();

    float sm = 0.0f;
    for (int j = tid; j < KKVV; j += 256) sm += __expf(gm[j] - mx);
    red[tid] = sm; __syncthreads();
    for (int s = 128; s > 0; s >>= 1) {
        if (tid < s) red[tid] += red[tid + s];
        __syncthreads();
    }
    const float inv = 1.0f / red[0];

    float a0 = 0.f, a1 = 0.f, a2 = 0.f;      // 768 = 3 * 256 channels
    for (int j = 0; j < KKVV; ++j) {
        const float w = __expf(gm[j] - mx);
        const __bf16* vrow = vg + ((size_t)b * KKVV + j) * HH;
        a0 += w * (float)vrow[tid];
        a1 += w * (float)vrow[tid + 256];
        a2 += w * (float)vrow[tid + 512];
    }
    bg[b * HH + tid]       = a0 * inv;
    bg[b * HH + tid + 256] = a1 * inv;
    bg[b * HH + tid + 512] = a2 * inv;
}

__global__ void k_fill(const float* __restrict__ bg, float* __restrict__ out) {
    size_t i = (size_t)blockIdx.x * blockDim.x + threadIdx.x;
    if (i >= (size_t)BB * TT * HH) return;
    int hh = (int)(i % HH);
    int b  = (int)(i / ((size_t)TT * HH));
    out[i] = bg[b * HH + hh];
}

// ---------------- flash attention over the 1024 selected kv tokens ----------------
// grid: (KQ/16, NH, B), block: 32 (1 wave). Online softmax over kv chunks of 32.
__global__ __launch_bounds__(32)
void k_attn(const __bf16* __restrict__ qg, const __bf16* __restrict__ kg,
            const __bf16* __restrict__ vt, const float* __restrict__ gmask,
            const int* __restrict__ qidx, float* __restrict__ out) {
    __shared__ float Plds[16 * 32];

    const int lane = threadIdx.x;
    const int c = lane & 15, half = lane >> 4;
    const int qt = blockIdx.x, h = blockIdx.y, b = blockIdx.z;

    // Q tile as two 16x32 A operands (K-dim = DH = 64)
    const int qrow = qt * 16 + c;
    const __bf16* qp = qg + ((size_t)b * KQQ + qrow) * HH + h * DHH;
    const v16bf a0 = load_a_chunk(qp, 0, half);
    const v16bf a1 = load_a_chunk(qp, 32, half);

    float m[8], l[8];
#pragma unroll
    for (int r = 0; r < 8; ++r) { m[r] = -3.0e38f; l[r] = 0.0f; }
    v8f O[4];
#pragma unroll
    for (int t = 0; t < 4; ++t) O[t] = (v8f)(0.0f);

    const float* gm = gmask + b * KKVV;
    const __bf16* kbase = kg + (size_t)b * KKVV * HH + h * DHH;
    const __bf16* vtb = vt + ((size_t)(b * NHH + h)) * DHH * KKVV;

    for (int j = 0; j < KKVV; j += 32) {
        // ---- scores: S0 = Q*K^T (kv j..j+15), S1 (kv j+16..j+31) ----
        v8f S0 = (v8f)(0.0f), S1 = (v8f)(0.0f);
        {
            const __bf16* kp = kbase + (size_t)(j + c) * HH + 16 * half;
            v16bf b00 = *(const v16bf*)kp;
            v16bf b01 = *(const v16bf*)(kp + 32);
            S0 = wmma_bf16(a0, b00, S0);
            S0 = wmma_bf16(a1, b01, S0);
        }
        {
            const __bf16* kp = kbase + (size_t)(j + 16 + c) * HH + 16 * half;
            v16bf b10 = *(const v16bf*)kp;
            v16bf b11 = *(const v16bf*)(kp + 32);
            S1 = wmma_bf16(a0, b10, S1);
            S1 = wmma_bf16(a1, b11, S1);
        }
        const float mk0 = gm[j + c];
        const float mk1 = gm[j + 16 + c];

        // ---- online softmax (rows r+8*half live in this lane's 16-lane group) ----
        float P0[8], P1[8];
#pragma unroll
        for (int r = 0; r < 8; ++r) {
            float s0 = S0[r] * 0.125f + mk0;       // 1/sqrt(64)
            float s1 = S1[r] * 0.125f + mk1;
            float cm = fmaxf(s0, s1);
#pragma unroll
            for (int off = 1; off < 16; off <<= 1)
                cm = fmaxf(cm, __shfl_xor(cm, off, 32));
            const float mn = fmaxf(m[r], cm);
            const float corr = __expf(m[r] - mn);
            const float p0 = __expf(s0 - mn);
            const float p1 = __expf(s1 - mn);
            float rs = p0 + p1;
#pragma unroll
            for (int off = 1; off < 16; off <<= 1)
                rs += __shfl_xor(rs, off, 32);
            l[r] = l[r] * corr + rs;
            m[r] = mn;
            O[0][r] *= corr; O[1][r] *= corr; O[2][r] *= corr; O[3][r] *= corr;
            P0[r] = p0; P1[r] = p1;
        }

        // ---- transpose P (C/D layout -> A layout) through LDS ----
        __syncthreads();
#pragma unroll
        for (int r = 0; r < 8; ++r) {
            Plds[(r + 8 * half) * 32 + c]      = P0[r];
            Plds[(r + 8 * half) * 32 + 16 + c] = P1[r];
        }
        __syncthreads();
        v16bf ap;
        const float* prow = Plds + c * 32;         // A row M = c
#pragma unroll
        for (int e = 0; e < 8; ++e) ap[e]     = (__bf16)prow[16 * half + e];
#pragma unroll
        for (int e = 0; e < 8; ++e) ap[8 + e] = (__bf16)prow[16 * half + 8 + e];

        // ---- ctx: O += P(16x32) * V(32x16) per 16-wide d tile ----
#pragma unroll
        for (int t = 0; t < 4; ++t) {
            const __bf16* vp = vtb + (size_t)(t * 16 + c) * KKVV + j + 16 * half;
            v16bf bvv = *(const v16bf*)vp;
            O[t] = wmma_bf16(ap, bvv, O[t]);
        }
    }

    // ---- epilogue: normalize, scatter to output tokens ----
    const int* qi = qidx + b * KQQ;
#pragma unroll
    for (int r = 0; r < 8; ++r) {
        const float inv = 1.0f / l[r];
        const int tokr = qi[qt * 16 + r + 8 * half];
        float* orow = out + ((size_t)b * TT + tokr) * HH + h * DHH;
#pragma unroll
        for (int t = 0; t < 4; ++t) orow[t * 16 + c] = O[t][r] * inv;
    }
}

// ---------------- launch ----------------
extern "C" void kernel_launch(void* const* d_in, const int* in_sizes, int n_in,
                              void* d_out, int out_size, void* d_ws, size_t ws_size,
                              hipStream_t stream) {
    const float* hidden = (const float*)d_in[0];
    const float* amask  = (const float*)d_in[1];
    const float* Wq = (const float*)d_in[2];
    const float* bq = (const float*)d_in[3];
    const float* Wk = (const float*)d_in[4];
    const float* bk = (const float*)d_in[5];
    const float* Wv = (const float*)d_in[6];
    const float* bv = (const float*)d_in[7];
    const int* qidx  = (const int*)d_in[8];
    const int* kvidx = (const int*)d_in[9];
    float* out = (float*)d_out;

    char* ws = (char*)d_ws;
    const size_t SZ_HID = (size_t)BB * TT * HH * 2;     // 12.6 MB bf16 hidden
    const size_t SZ_W   = (size_t)HH * HH * 2;          // 1.18 MB per weight
    const size_t SZ_G   = (size_t)BB * KQQ * HH * 2;    // 6.3 MB per gathered proj
    __bf16* hid_bf = (__bf16*)(ws);
    __bf16* wq_bf  = (__bf16*)(ws + SZ_HID);
    __bf16* wk_bf  = (__bf16*)(ws + SZ_HID + SZ_W);
    __bf16* wv_bf  = (__bf16*)(ws + SZ_HID + 2 * SZ_W);
    __bf16* qg     = (__bf16*)(ws + SZ_HID + 3 * SZ_W);
    __bf16* kg     = (__bf16*)(ws + SZ_HID + 3 * SZ_W + SZ_G);
    __bf16* vg     = (__bf16*)(ws + SZ_HID + 3 * SZ_W + 2 * SZ_G);
    __bf16* vt     = (__bf16*)(ws + SZ_HID + 3 * SZ_W + 3 * SZ_G);
    float* gmaskp  = (float*)(ws + SZ_HID + 3 * SZ_W + 4 * SZ_G);
    float* bgp     = (float*)(ws + SZ_HID + 3 * SZ_W + 4 * SZ_G + (size_t)BB * KKVV * 4);
    // total ~41.3 MB of ws

    const int nh = BB * TT * HH;
    k_cvt_bf16<<<(nh + 255) / 256, 256, 0, stream>>>(hidden, hid_bf, nh);
    k_cvt_bf16<<<(HH * HH + 255) / 256, 256, 0, stream>>>(Wq, wq_bf, HH * HH);
    k_cvt_bf16<<<(HH * HH + 255) / 256, 256, 0, stream>>>(Wk, wk_bf, HH * HH);
    k_cvt_bf16<<<(HH * HH + 255) / 256, 256, 0, stream>>>(Wv, wv_bf, HH * HH);
    k_gmask<<<(BB * KKVV + 255) / 256, 256, 0, stream>>>(amask, kvidx, gmaskp);

    k_proj<<<dim3(KQQ / 16, HH / 64, BB), 32, 0, stream>>>(hid_bf, wq_bf, bq, qidx, qg, KQQ);
    k_proj<<<dim3(KKVV / 16, HH / 64, BB), 32, 0, stream>>>(hid_bf, wk_bf, bk, kvidx, kg, KKVV);
    k_proj<<<dim3(KKVV / 16, HH / 64, BB), 32, 0, stream>>>(hid_bf, wv_bf, bv, kvidx, vg, KKVV);

    k_vtrans<<<(BB * KKVV * HH + 255) / 256, 256, 0, stream>>>(vg, vt);
    k_bg<<<BB, 256, 0, stream>>>(gmaskp, vg, bgp);
    k_fill<<<(nh + 255) / 256, 256, 0, stream>>>(bgp, out);

    k_attn<<<dim3(KQQ / 16, NHH, BB), 32, 0, stream>>>(qg, kg, vt, gmaskp, qidx, out);
}
